// SPN_29935922053487
// MI455X (gfx1250) — matrix operand order repaired and checked by reading, not compiled, not run on hardware
//
#include <hip/hip_runtime.h>
#include <hip/hip_bf16.h>

typedef __bf16 bh;
typedef __attribute__((ext_vector_type(16))) __bf16 v16bf;
typedef __attribute__((ext_vector_type(8)))  __bf16 v8bf;
typedef __attribute__((ext_vector_type(4)))  __bf16 v4bf;
typedef __attribute__((ext_vector_type(8)))  float  v8f;

#define N_NODES 100000
#define H_DIM   128
#define E_EDGES 3200000
#define K_DIST  5
#define C_CLS   40
#define C_PAD   48

// ---------------------------------------------------------------------------
// Utility kernels
// ---------------------------------------------------------------------------
__global__ void k_zero4(float4* __restrict__ p, int n4) {
    int i = blockIdx.x * blockDim.x + threadIdx.x;
    if (i < n4) p[i] = make_float4(0.f, 0.f, 0.f, 0.f);
}

__global__ void k_f2bf(const float4* __restrict__ src, v4bf* __restrict__ dst, int n4) {
    int i = blockIdx.x * blockDim.x + threadIdx.x;
    if (i >= n4) return;
    float4 v = src[i];
    v4bf o;
    o[0] = (bh)v.x; o[1] = (bh)v.y; o[2] = (bh)v.z; o[3] = (bh)v.w;
    dst[i] = o;
}

// Transpose-convert weight W[k][Nout] (row-major over k) -> Wt[n][k] bf16,
// zero-padding columns n in [Nout, NoutPad).
__global__ void k_wconv(const float* __restrict__ W, bh* __restrict__ Wt,
                        int Nout, int NoutPad) {
    int idx = blockIdx.x * blockDim.x + threadIdx.x;
    if (idx >= NoutPad * H_DIM) return;
    int n = idx >> 7;          // output-feature index
    int k = idx & 127;         // reduction index
    float v = (n < Nout) ? W[(size_t)k * Nout + n] : 0.0f;
    Wt[idx] = (bh)v;
}

// Softmax over the two K=5 hop-weight vectors -> w[0..4], w[5..9]
__global__ void k_softmax2(const float* __restrict__ hc0,
                           const float* __restrict__ hc1,
                           float* __restrict__ w) {
    if (threadIdx.x != 0 || blockIdx.x != 0) return;
    for (int j = 0; j < 2; ++j) {
        const float* hc = j ? hc1 : hc0;
        float mx = hc[0];
        for (int i = 1; i < K_DIST; ++i) mx = fmaxf(mx, hc[i]);
        float e[K_DIST], s = 0.f;
        for (int i = 0; i < K_DIST; ++i) { e[i] = __expf(hc[i] - mx); s += e[i]; }
        float inv = 1.f / s;
        for (int i = 0; i < K_DIST; ++i) w[j * K_DIST + i] = e[i] * inv;
    }
}

// ---------------------------------------------------------------------------
// WMMA GEMM: Z[m][n] = sum_k A_bf16[m][k] * Wt_bf16[n][k] + bias[n]
// One wave = one 16x16 fp32 tile, K=128 via 4x v_wmma_f32_16x16x32_bf16.
// Also accumulates per-column sum / sum-of-squares for BatchNorm (atomics).
// ---------------------------------------------------------------------------
__global__ __launch_bounds__(256) void k_gemm(
    const bh* __restrict__ A,      // [Nrows,128] bf16 row-major
    const bh* __restrict__ Wt,     // [NoutPad,128] bf16 (pre-transposed)
    const float* __restrict__ bias,// [NoutReal]
    float* __restrict__ Z,         // [Nrows,NoutReal] fp32
    float* __restrict__ stats,     // [2*NoutReal] col sum / sumsq, or nullptr
    int Nrows, int NoutReal)
{
    const int wave = threadIdx.x >> 5;
    const int lane = threadIdx.x & 31;
    const int m0 = (blockIdx.x * 8 + wave) * 16;
    if (m0 >= Nrows) return;                 // wave-uniform: EXEC stays all-ones
    const int n0 = blockIdx.y * 16;
    const int lr = lane & 15;
    const int hi = lane >> 4;                // lane group 0/1

    // A layout (16-bit A 16x32): lanes 0-15 -> K {0..7,16..23}; 16-31 -> {8..15,24..31}
    const bh* arow = A  + (size_t)(m0 + lr) * H_DIM + hi * 8;
    // B layout (16-bit B 32x16): lanes 0-15 -> K {0..7,16..23}; 16-31 -> {8..15,24..31}
    const bh* brow = Wt + (size_t)(n0 + lr) * H_DIM + hi * 8;

    v8f acc = {};
#pragma unroll
    for (int kt = 0; kt < 4; ++kt) {
        v8bf a0 = *(const v8bf*)(arow + kt * 32);
        v8bf a1 = *(const v8bf*)(arow + kt * 32 + 16);
        v8bf b0 = *(const v8bf*)(brow + kt * 32);
        v8bf b1 = *(const v8bf*)(brow + kt * 32 + 16);
        v16bf a, b;
#pragma unroll
        for (int i = 0; i < 8; ++i) {
            a[i] = a0[i]; a[i + 8] = a1[i];
            b[i] = b0[i]; b[i + 8] = b1[i];
        }
        acc = __builtin_amdgcn_wmma_f32_16x16x32_bf16(
            /*neg_a=*/false, a, /*neg_b=*/false, b,
            /*c_mod=*/(short)0, acc, /*reuse_a=*/false, /*reuse_b=*/false);
    }

    const int n = n0 + lr;
    const float bv = (n < NoutReal) ? bias[n] : 0.0f;
    float s = 0.f, ss = 0.f;
#pragma unroll
    for (int r = 0; r < 8; ++r) {
        float v = acc[r] + bv;
        acc[r] = v; s += v; ss += v * v;
    }
    if (n < NoutReal) {
        const int mbase = m0 + hi * 8;       // C/D layout: lanes 16-31 -> rows M+8
#pragma unroll
        for (int r = 0; r < 8; ++r)
            Z[(size_t)(mbase + r) * NoutReal + n] = acc[r];
        if (stats) {
            atomicAdd(&stats[n], s);
            atomicAdd(&stats[NoutReal + n], ss);
        }
    }
}

// Finalize BN: scale = g * rsqrt(var+eps), shift = beta - scale*mean  (H=128)
__global__ void k_bnfinal(const float* __restrict__ stats,
                          const float* __restrict__ g,
                          const float* __restrict__ bt,
                          float* __restrict__ scsh, float invN) {
    int n = threadIdx.x;                      // launched with 128 threads
    float m = stats[n] * invN;
    float v = stats[H_DIM + n] * invN - m * m;
    float sc = g[n] * rsqrtf(v + 1e-5f);
    scsh[n] = sc;
    scsh[H_DIM + n] = bt[n] - sc * m;
}

// Apply BN + ReLU; write optional fp32 master and bf16 shadow for next GEMM.
__global__ void k_bnapply(const float4* __restrict__ Z,
                          const float* __restrict__ scsh,
                          float* __restrict__ Hf32,      // may be nullptr
                          v4bf* __restrict__ Abf, int n4) {
    int i = blockIdx.x * blockDim.x + threadIdx.x;
    if (i >= n4) return;
    float4 z = Z[i];
    int c = (i * 4) & 127;
    float4 y;
    y.x = fmaxf(0.f, scsh[c    ] * z.x + scsh[H_DIM + c    ]);
    y.y = fmaxf(0.f, scsh[c + 1] * z.y + scsh[H_DIM + c + 1]);
    y.z = fmaxf(0.f, scsh[c + 2] * z.z + scsh[H_DIM + c + 2]);
    y.w = fmaxf(0.f, scsh[c + 3] * z.w + scsh[H_DIM + c + 3]);
    if (Hf32) ((float4*)Hf32)[i] = y;
    v4bf o;
    o[0] = (bh)y.x; o[1] = (bh)y.y; o[2] = (bh)y.z; o[3] = (bh)y.w;
    Abf[i] = o;
}

// Edge scatter: agg[dst] += w[dist] * h_bf16[src]; one wave per edge,
// 4 features per lane, fp32 atomics into L2-resident agg.
__global__ __launch_bounds__(256) void k_scatter(
    const bh* __restrict__ Hbf, const int* __restrict__ ei,
    const int* __restrict__ dist, const float* __restrict__ wsoft,
    float* __restrict__ agg, int E)
{
    int e = blockIdx.x * 8 + (threadIdx.x >> 5);
    if (e >= E) return;
    int lane = threadIdx.x & 31;
    int src = ei[e];
    int dst = ei[E + e];
    float w = wsoft[dist[e]];
    v4bf h = *(const v4bf*)(Hbf + (size_t)src * H_DIM + lane * 4);
    float* a = agg + (size_t)dst * H_DIM + lane * 4;
    atomicAdd(a + 0, w * (float)h[0]);
    atomicAdd(a + 1, w * (float)h[1]);
    atomicAdd(a + 2, w * (float)h[2]);
    atomicAdd(a + 3, w * (float)h[3]);
}

// GIN combine: Abf = bf16(h + agg)   (eps = 0)
__global__ void k_combine(const float4* __restrict__ H,
                          const float4* __restrict__ agg,
                          v4bf* __restrict__ Abf, int n4) {
    int i = blockIdx.x * blockDim.x + threadIdx.x;
    if (i >= n4) return;
    float4 a = H[i], b = agg[i];
    v4bf o;
    o[0] = (bh)(a.x + b.x); o[1] = (bh)(a.y + b.y);
    o[2] = (bh)(a.z + b.z); o[3] = (bh)(a.w + b.w);
    Abf[i] = o;
}

// ---------------------------------------------------------------------------
extern "C" void kernel_launch(void* const* d_in, const int* in_sizes, int n_in,
                              void* d_out, int out_size, void* d_ws, size_t ws_size,
                              hipStream_t stream) {
    const float* x    = (const float*)d_in[0];
    const int*   ei   = (const int*)d_in[1];
    const int*   dist = (const int*)d_in[2];
    const float* Wi  = (const float*)d_in[3];
    const float* bi  = (const float*)d_in[4];
    const float* gi  = (const float*)d_in[5];
    const float* bti = (const float*)d_in[6];
    const float* hc0 = (const float*)d_in[7];
    const float* W0a = (const float*)d_in[8],  *b0a = (const float*)d_in[9];
    const float* g0a = (const float*)d_in[10], *bt0a = (const float*)d_in[11];
    const float* W0b = (const float*)d_in[12], *b0b = (const float*)d_in[13];
    const float* g0b = (const float*)d_in[14], *bt0b = (const float*)d_in[15];
    const float* hc1 = (const float*)d_in[16];
    const float* W1a = (const float*)d_in[17], *b1a = (const float*)d_in[18];
    const float* g1a = (const float*)d_in[19], *bt1a = (const float*)d_in[20];
    const float* W1b = (const float*)d_in[21], *b1b = (const float*)d_in[22];
    const float* g1b = (const float*)d_in[23], *bt1b = (const float*)d_in[24];
    const float* Wp1 = (const float*)d_in[25], *bp1 = (const float*)d_in[26];
    const float* gp1 = (const float*)d_in[27], *btp1 = (const float*)d_in[28];
    const float* Wp2 = (const float*)d_in[29], *bp2 = (const float*)d_in[30];

    const size_t NH = (size_t)N_NODES * H_DIM;        // 12.8M
    const int    n4 = (int)(NH / 4);                  // float4 / v4bf count

    // Workspace carve (256B aligned)
    char* p = (char*)d_ws;
    auto carve = [&](size_t bytes) -> char* {
        char* r = p; p += (bytes + 255) & ~(size_t)255; return r;
    };
    float* Z    = (float*)carve(NH * 4);   // GEMM out; aliased with agg (disjoint lifetimes)
    float* Hf   = (float*)carve(NH * 4);   // fp32 master activation
    bh*    Abf  = (bh*)   carve(NH * 2);   // bf16 shadow (GEMM A input / scatter source)
    bh*    WtWi  = (bh*)carve(H_DIM * H_DIM * 2);
    bh*    WtW0a = (bh*)carve(H_DIM * H_DIM * 2);
    bh*    WtW0b = (bh*)carve(H_DIM * H_DIM * 2);
    bh*    WtW1a = (bh*)carve(H_DIM * H_DIM * 2);
    bh*    WtW1b = (bh*)carve(H_DIM * H_DIM * 2);
    bh*    WtWp1 = (bh*)carve(H_DIM * H_DIM * 2);
    bh*    WtWp2 = (bh*)carve(C_PAD * H_DIM * 2);
    float* stats = (float*)carve(256 * 4);
    float* scsh  = (float*)carve(256 * 4);
    float* wsoft = (float*)carve(64);
    float* agg   = Z;                      // alias

    // --- weight transpose+convert (tiny) ---
    const int wthreads = 256;
    const int wgrid128 = (H_DIM * H_DIM + wthreads - 1) / wthreads;
    k_wconv<<<wgrid128, wthreads, 0, stream>>>(Wi,  WtWi,  H_DIM, H_DIM);
    k_wconv<<<wgrid128, wthreads, 0, stream>>>(W0a, WtW0a, H_DIM, H_DIM);
    k_wconv<<<wgrid128, wthreads, 0, stream>>>(W0b, WtW0b, H_DIM, H_DIM);
    k_wconv<<<wgrid128, wthreads, 0, stream>>>(W1a, WtW1a, H_DIM, H_DIM);
    k_wconv<<<wgrid128, wthreads, 0, stream>>>(W1b, WtW1b, H_DIM, H_DIM);
    k_wconv<<<wgrid128, wthreads, 0, stream>>>(Wp1, WtWp1, H_DIM, H_DIM);
    k_wconv<<<(C_PAD * H_DIM + wthreads - 1) / wthreads, wthreads, 0, stream>>>(
        Wp2, WtWp2, C_CLS, C_PAD);
    k_softmax2<<<1, 32, 0, stream>>>(hc0, hc1, wsoft);

    // --- x -> bf16 ---
    const int egrid = (n4 + 255) / 256;
    k_f2bf<<<egrid, 256, 0, stream>>>((const float4*)x, (v4bf*)Abf, n4);

    const dim3 gemmGrid((N_NODES / 16 + 7) / 8, H_DIM / 16);  // 782 x 8
    const float invN = 1.0f / (float)N_NODES;

    auto mlp = [&](const bh* Wt_, const float* b_, const float* g_,
                   const float* bt_, bool keepH) {
        k_zero4<<<1, 64, 0, stream>>>((float4*)stats, 64);
        k_gemm<<<gemmGrid, 256, 0, stream>>>(Abf, Wt_, b_, Z, stats, N_NODES, H_DIM);
        k_bnfinal<<<1, H_DIM, 0, stream>>>(stats, g_, bt_, scsh, invN);
        k_bnapply<<<egrid, 256, 0, stream>>>((const float4*)Z, scsh,
                                             keepH ? Hf : nullptr, (v4bf*)Abf, n4);
    };
    auto spn_agg = [&](const float* wsub) {
        k_zero4<<<egrid, 256, 0, stream>>>((float4*)agg, n4);
        k_scatter<<<(E_EDGES + 7) / 8, 256, 0, stream>>>(Abf, ei, dist, wsub, agg, E_EDGES);
        k_combine<<<egrid, 256, 0, stream>>>((const float4*)Hf, (const float4*)agg,
                                             (v4bf*)Abf, n4);
    };

    // initial MLP: h0 = relu(bn(x @ Wi + bi))
    mlp(WtWi, bi, gi, bti, /*keepH=*/true);
    // SPN layer 0
    spn_agg(wsoft);
    mlp(WtW0a, b0a, g0a, bt0a, false);
    mlp(WtW0b, b0b, g0b, bt0b, true);
    // SPN layer 1
    spn_agg(wsoft + K_DIST);
    mlp(WtW1a, b1a, g1a, bt1a, false);
    mlp(WtW1b, b1b, g1b, bt1b, true);
    // head
    mlp(WtWp1, bp1, gp1, btp1, false);
    const dim3 headGrid((N_NODES / 16 + 7) / 8, C_PAD / 16);  // 782 x 3
    k_gemm<<<headGrid, 256, 0, stream>>>(Abf, WtWp2, bp2, (float*)d_out,
                                         nullptr, N_NODES, C_CLS);
}
